// MultiDoubleStreamBlockLoraProcessor_62319975465670
// MI455X (gfx1250) — compile-verified, matrix operand order checked by reading, not run on hardware
//
#include <hip/hip_runtime.h>
#include <hip/hip_bf16.h>

// ---------------- problem constants ----------------
#define DIM   3072
#define HEADS 24
#define HD    128
#define S_H   3072
#define S_E   512
#define S_ALL 3584
#define COND  1024
#define RANK  64
#define NEGBIG (-1.0e20f)

typedef __attribute__((ext_vector_type(16))) __bf16 v16bf;
typedef __attribute__((ext_vector_type(4)))  __bf16 v4bf;
typedef __attribute__((ext_vector_type(8)))  float  v8f;
typedef __attribute__((ext_vector_type(4)))  int    v4i;

// ---- async global->LDS (CDNA5): use if the toolchain declares it ----
#if defined(__HIP_DEVICE_COMPILE__) && __has_builtin(__builtin_amdgcn_global_load_async_to_lds_b128)
#define USE_ASYNC_LDS 1
typedef __attribute__((address_space(1))) v4i* as1_v4i_ptr;
typedef __attribute__((address_space(3))) v4i* as3_v4i_ptr;
#else
#define USE_ASYNC_LDS 0
#endif

__device__ __forceinline__ void copy_b128_gl(const __bf16* g, __bf16* l) {
#if USE_ASYNC_LDS
  __builtin_amdgcn_global_load_async_to_lds_b128((as1_v4i_ptr)g, (as3_v4i_ptr)l, 0, 0);
#else
  *(uint4*)l = *(const uint4*)g;
#endif
}

__device__ __forceinline__ void wait_async_all() {
#if USE_ASYNC_LDS
#if __has_builtin(__builtin_amdgcn_s_wait_asynccnt)
  __builtin_amdgcn_s_wait_asynccnt(0);
#else
  asm volatile("s_wait_asynccnt 0x0" ::: "memory");
#endif
#endif
}

union FragU { uint4 u[2]; v16bf v; };

// Build a 16x32 bf16 WMMA A/B fragment from an LDS row pointer.
// Per ISA layout: lane half (lane>=16) selects K-offsets {8*half} and {16+8*half}.
__device__ __forceinline__ v16bf ld_frag(const __bf16* rowptr, int kbase, int half) {
  FragU t;
  t.u[0] = *(const uint4*)(rowptr + kbase + 8 * half);
  t.u[1] = *(const uint4*)(rowptr + kbase + 16 + 8 * half);
  return t.v;
}

__device__ __forceinline__ v8f wmma_bf16(v16bf a, v16bf b, v8f c) {
  return __builtin_amdgcn_wmma_f32_16x16x32_bf16(false, a, false, b, (short)0, c, false, false);
}

// ---------------- fp32 -> bf16 conversion (vectorized, bandwidth-bound) ----------------
__global__ void f32_to_bf16_kernel(const float* __restrict__ in, __bf16* __restrict__ out, long long n) {
  long long i = ((long long)blockIdx.x * blockDim.x + threadIdx.x) * 4;
  if (i + 3 < n) {
    float4 f = *(const float4*)(in + i);
    v4bf o;
    o.x = (__bf16)f.x; o.y = (__bf16)f.y; o.z = (__bf16)f.z; o.w = (__bf16)f.w;
    *(v4bf*)(out + i) = o;
  } else {
    for (; i < n; ++i) out[i] = (__bf16)in[i];
  }
}

// ---------------- generic bf16 WMMA GEMM, double-buffered LDS ----------------
// C[M,N] = (accumulate? C : 0) + scale*(*scale_ptr) * A[M,K] @ B[N,K]^T + bias[N]
#define BM 128
#define BN 128
#define BK 32

template<bool GUARD>
__global__ __launch_bounds__(256)
void gemm_bf16_kernel(const __bf16* __restrict__ A, int lda,
                      const __bf16* __restrict__ B, int ldb,
                      float* __restrict__ C, int ldc,
                      const float* __restrict__ bias,
                      int M, int N, int K,
                      float scale, const float* __restrict__ scale_ptr, int accumulate)
{
  __shared__ __attribute__((aligned(16))) __bf16 As[2][BM][BK];
  __shared__ __attribute__((aligned(16))) __bf16 Bs[2][BM][BK];

  int tid  = threadIdx.x;
  int wave = tid >> 5;
  int lane = tid & 31;
  int half = lane >> 4;
  int l16  = lane & 15;
  int wm   = wave & 1;   // 2 waves in M (64 rows each)
  int wn   = wave >> 1;  // 4 waves in N (32 cols each)

  int m_base = blockIdx.y * BM;
  int n_base = blockIdx.x * BN;

  int lrow[2], lch[2];
  #pragma unroll
  for (int j = 0; j < 2; ++j) {
    int idx = tid + 256 * j;
    lrow[j] = idx >> 2;   // 4 uint4 chunks per 32-element row
    lch[j]  = idx & 3;
  }

  // Fill one LDS tile pair. Unguarded: async direct global->LDS (no VGPR round
  // trip, tracked on ASYNCcnt). Guarded: reg-staged with zero fill.
  auto fill_tile = [&](int buf, int k0) {
    #pragma unroll
    for (int j = 0; j < 2; ++j) {
      if (GUARD) {
        uint4 va = make_uint4(0u, 0u, 0u, 0u);
        uint4 vb = make_uint4(0u, 0u, 0u, 0u);
        if (m_base + lrow[j] < M)
          va = *(const uint4*)(A + (size_t)(m_base + lrow[j]) * lda + k0 + lch[j] * 8);
        if (n_base + lrow[j] < N)
          vb = *(const uint4*)(B + (size_t)(n_base + lrow[j]) * ldb + k0 + lch[j] * 8);
        *(uint4*)(&As[buf][lrow[j]][lch[j] * 8]) = va;
        *(uint4*)(&Bs[buf][lrow[j]][lch[j] * 8]) = vb;
      } else {
        copy_b128_gl(A + (size_t)(m_base + lrow[j]) * lda + k0 + lch[j] * 8,
                     &As[buf][lrow[j]][lch[j] * 8]);
        copy_b128_gl(B + (size_t)(n_base + lrow[j]) * ldb + k0 + lch[j] * 8,
                     &Bs[buf][lrow[j]][lch[j] * 8]);
      }
    }
  };

  v8f acc[4][2] = {};

  fill_tile(0, 0);
  if (!GUARD) wait_async_all();
  __syncthreads();

  int buf = 0;
  for (int k0 = 0; k0 < K; k0 += BK) {
    bool has_next = (k0 + BK) < K;
    if (has_next) fill_tile(buf ^ 1, k0 + BK);  // copies in flight during WMMA

    v16bf bfrag[2];
    #pragma unroll
    for (int ns = 0; ns < 2; ++ns)
      bfrag[ns] = ld_frag(&Bs[buf][wn * 32 + ns * 16 + l16][0], 0, half);

    #pragma unroll
    for (int ms = 0; ms < 4; ++ms) {
      v16bf afrag = ld_frag(&As[buf][wm * 64 + ms * 16 + l16][0], 0, half);
      #pragma unroll
      for (int ns = 0; ns < 2; ++ns)
        acc[ms][ns] = wmma_bf16(afrag, bfrag[ns], acc[ms][ns]);
    }

    if (!GUARD && has_next) wait_async_all();   // after the WMMA block
    __syncthreads();
    buf ^= 1;
  }

  float smul = scale * (scale_ptr ? *scale_ptr : 1.0f);
  #pragma unroll
  for (int ms = 0; ms < 4; ++ms) {
    #pragma unroll
    for (int ns = 0; ns < 2; ++ns) {
      int col = n_base + wn * 32 + ns * 16 + l16;
      if (GUARD && col >= N) continue;
      #pragma unroll
      for (int r = 0; r < 8; ++r) {
        int row = m_base + wm * 64 + ms * 16 + r + 8 * half;
        if (GUARD && row >= M) continue;
        float v = acc[ms][ns][r] * smul;
        if (bias) v += bias[col];
        size_t o = (size_t)row * ldc + col;
        if (accumulate) v += C[o];
        C[o] = v;
      }
    }
  }
}

// ---------------- heads + RMSNorm + RoPE + concat pack ----------------
// Produces Q,K,V as bf16 [HEADS][S_ALL][HD].
__global__ __launch_bounds__(128)
void qkv_pack_kernel(const float* __restrict__ q_hid, const float* __restrict__ k_hid,
                     const float* __restrict__ v_hid,
                     const float* __restrict__ q_enc, const float* __restrict__ k_enc,
                     const float* __restrict__ v_enc,
                     const float* __restrict__ nq, const float* __restrict__ nk,
                     const float* __restrict__ naq, const float* __restrict__ nak,
                     const float* __restrict__ rope_cos, const float* __restrict__ rope_sin,
                     __bf16* __restrict__ Qb, __bf16* __restrict__ Kb, __bf16* __restrict__ Vb)
{
  __shared__ float red[4];
  int s = blockIdx.x;
  int h = blockIdx.y;
  int d = threadIdx.x;

  const float *sq, *sk, *sv, *wq, *wk;
  int row;
  if (s < S_E) { sq = q_enc; sk = k_enc; sv = v_enc; wq = naq; wk = nak; row = s; }
  else         { sq = q_hid; sk = k_hid; sv = v_hid; wq = nq;  wk = nk;  row = s - S_E; }

  size_t base = (size_t)row * DIM + h * HD + d;
  float qv = sq[base];
  float kv = sk[base];
  float vv = sv[base];

  // RMS over the 128 dims of this head (4 waves -> LDS reduce)
  auto rms = [&](float x, const float* w) -> float {
    float ss = x * x;
    #pragma unroll
    for (int off = 16; off > 0; off >>= 1) ss += __shfl_xor(ss, off);
    int wv = threadIdx.x >> 5;
    if ((threadIdx.x & 31) == 0) red[wv] = ss;
    __syncthreads();
    float tot = red[0] + red[1] + red[2] + red[3];
    __syncthreads();
    return x * rsqrtf(tot / (float)HD + 1e-6f) * w[d];
  };

  float qn = rms(qv, wq);
  float kn = rms(kv, wk);

  float c  = rope_cos[(size_t)s * HD + d];
  float sn = rope_sin[(size_t)s * HD + d];
  // interleaved pairs: even d uses -x[d+1], odd d uses x[d-1]
  float qo = __shfl_xor(qn, 1);
  float ko = __shfl_xor(kn, 1);
  float qr = (d & 1) ? qo : -qo;
  float kr = (d & 1) ? ko : -ko;

  size_t o = ((size_t)h * S_ALL + s) * HD + d;
  Qb[o] = (__bf16)(qn * c + qr * sn);
  Kb[o] = (__bf16)(kn * c + kr * sn);
  Vb[o] = (__bf16)vv;
}

// ---------------- flash attention (WMMA, bf16 in / fp32 softmax) ----------------
#define AT_BQ 64
#define AT_BK 64

__device__ __forceinline__ float mask_val(int q, int k) {
  if (q < S_E) return 0.0f;
  if (q < S_E + COND && k >= S_E && k < S_E + COND) return 0.0f;
  return NEGBIG;
}

__global__ __launch_bounds__(128)
void attn_kernel(const __bf16* __restrict__ Qg, const __bf16* __restrict__ Kg,
                 const __bf16* __restrict__ Vg, __bf16* __restrict__ O)
{
  __shared__ __attribute__((aligned(16))) __bf16 Qs[AT_BQ][HD];
  __shared__ __attribute__((aligned(16))) __bf16 Ks[AT_BK][HD];
  __shared__ __attribute__((aligned(16))) __bf16 Vt[HD][AT_BK];      // transposed V tile
  __shared__ __attribute__((aligned(16))) __bf16 Ps[4][16][AT_BK];   // per-wave P tile

  int h    = blockIdx.y;
  int q0   = blockIdx.x * AT_BQ;
  int tid  = threadIdx.x;
  int wave = tid >> 5;
  int lane = tid & 31;
  int half = lane >> 4;
  int l16  = lane & 15;

  const __bf16* Qh = Qg + (size_t)h * S_ALL * HD;
  const __bf16* Kh = Kg + (size_t)h * S_ALL * HD;
  const __bf16* Vh = Vg + (size_t)h * S_ALL * HD;

  // load Q tile (64x128 bf16 = 1024 b128 copies, 8 per thread)
  for (int i = tid; i < AT_BQ * HD / 8; i += 128) {
    int row = i >> 4;
    int ch  = i & 15;
    copy_b128_gl(Qh + (size_t)(q0 + row) * HD + ch * 8, &Qs[row][ch * 8]);
  }
  wait_async_all();

  v8f o[8] = {};
  float mrow[8], lrow[8];
  #pragma unroll
  for (int r = 0; r < 8; ++r) { mrow[r] = -3.0e38f; lrow[r] = 0.0f; }

  const float sm_scale = 0.088388347648318447f; // 1/sqrt(128)

  for (int kv0 = 0; kv0 < S_ALL; kv0 += AT_BK) {
    __syncthreads();
    for (int i = tid; i < AT_BK * HD / 8; i += 128) {
      int row = i >> 4;
      int ch  = i & 15;
      copy_b128_gl(Kh + (size_t)(kv0 + row) * HD + ch * 8, &Ks[row][ch * 8]);
    }
    // V tile transposed: b128 on the global side, 2B scatter into LDS
    for (int i = tid; i < AT_BK * HD / 8; i += 128) {
      int kv = i >> 4;
      int ch = i & 15;
      uint4 u = *(const uint4*)(Vh + (size_t)(kv0 + kv) * HD + ch * 8);
      const __bf16* e = (const __bf16*)&u;
      #pragma unroll
      for (int j = 0; j < 8; ++j) Vt[ch * 8 + j][kv] = e[j];
    }
    // warm L2 for the next tile while we compute this one
    if (kv0 + AT_BK < S_ALL) {
      __builtin_prefetch(Kh + (size_t)(kv0 + AT_BK) * HD + tid * 16, 0, 0);
      __builtin_prefetch(Vh + (size_t)(kv0 + AT_BK) * HD + tid * 16, 0, 0);
    }
    wait_async_all();
    __syncthreads();

    // ---- scores: S = Q K^T (16 rows per wave x 64 cols) ----
    v8f sc[4] = {};
    #pragma unroll
    for (int kc = 0; kc < 4; ++kc) {
      v16bf af = ld_frag(&Qs[wave * 16 + l16][0], kc * 32, half);
      #pragma unroll
      for (int ct = 0; ct < 4; ++ct) {
        v16bf bf_ = ld_frag(&Ks[ct * 16 + l16][0], kc * 32, half);
        sc[ct] = wmma_bf16(af, bf_, sc[ct]);
      }
    }

    // ---- scale + mask (fp32, reproduces the -1e20 absorption quirk) ----
    #pragma unroll
    for (int ct = 0; ct < 4; ++ct) {
      int kcol = kv0 + ct * 16 + l16;
      #pragma unroll
      for (int r = 0; r < 8; ++r) {
        int qrow = q0 + wave * 16 + r + 8 * half;
        sc[ct][r] = sc[ct][r] * sm_scale + mask_val(qrow, kcol);
      }
    }

    // ---- online softmax update ----
    #pragma unroll
    for (int r = 0; r < 8; ++r) {
      float m = fmaxf(fmaxf(sc[0][r], sc[1][r]), fmaxf(sc[2][r], sc[3][r]));
      #pragma unroll
      for (int off = 1; off < 16; off <<= 1) m = fmaxf(m, __shfl_xor(m, off));
      float mnew = fmaxf(mrow[r], m);
      float corr = __expf(mrow[r] - mnew);
      mrow[r] = mnew;
      lrow[r] *= corr;
      #pragma unroll
      for (int oc = 0; oc < 8; ++oc) o[oc][r] *= corr;
      float psum = 0.0f;
      #pragma unroll
      for (int ct = 0; ct < 4; ++ct) {
        float p = __expf(sc[ct][r] - mnew);
        sc[ct][r] = p;
        psum += p;
      }
      #pragma unroll
      for (int off = 1; off < 16; off <<= 1) psum += __shfl_xor(psum, off);
      lrow[r] += psum;
    }

    // ---- stage P (bf16) in per-wave LDS for A-fragment layout ----
    #pragma unroll
    for (int ct = 0; ct < 4; ++ct)
      #pragma unroll
      for (int r = 0; r < 8; ++r)
        Ps[wave][r + 8 * half][ct * 16 + l16] = (__bf16)sc[ct][r];
    asm volatile("s_wait_dscnt 0" ::: "memory");

    // ---- O += P @ V ----
    #pragma unroll
    for (int kc = 0; kc < 2; ++kc) {
      v16bf af = ld_frag(&Ps[wave][l16][0], kc * 32, half);
      #pragma unroll
      for (int oc = 0; oc < 8; ++oc) {
        v16bf bf_ = ld_frag(&Vt[oc * 16 + l16][0], kc * 32, half);
        o[oc] = wmma_bf16(af, bf_, o[oc]);
      }
    }
  }

  // ---- finalize: write [s, h*HD + d] bf16 for the output projection ----
  #pragma unroll
  for (int r = 0; r < 8; ++r) {
    float inv = 1.0f / lrow[r];
    int srow = q0 + wave * 16 + r + 8 * half;
    #pragma unroll
    for (int oc = 0; oc < 8; ++oc) {
      int d = oc * 16 + l16;
      O[(size_t)srow * DIM + h * HD + d] = (__bf16)(o[oc][r] * inv);
    }
  }
}

// ---------------- host-side orchestration ----------------
extern "C" void kernel_launch(void* const* d_in, const int* in_sizes, int n_in,
                              void* d_out, int out_size, void* d_ws, size_t ws_size,
                              hipStream_t stream) {
  const float* hidden  = (const float*)d_in[0];
  const float* encoder = (const float*)d_in[1];
  const float* ropeC   = (const float*)d_in[2];
  const float* ropeS   = (const float*)d_in[3];
  const float* W[8];  for (int i = 0; i < 8; ++i) W[i]  = (const float*)d_in[4 + i];
  const float* bv[8]; for (int i = 0; i < 8; ++i) bv[i] = (const float*)d_in[12 + i];
  const float* norm_q  = (const float*)d_in[20];
  const float* norm_k  = (const float*)d_in[21];
  const float* norm_aq = (const float*)d_in[22];
  const float* norm_ak = (const float*)d_in[23];
  const float* down[3]; for (int i = 0; i < 3; ++i) down[i] = (const float*)d_in[24 + i];
  const float* up[3];   for (int i = 0; i < 3; ++i) up[i]   = (const float*)d_in[27 + i];
  const float* lora_w  = (const float*)d_in[30];

  float* out_hid = (float*)d_out;
  float* out_enc = (float*)d_out + (size_t)S_H * DIM;

  // workspace bump allocator (256B aligned)
  char* p = (char*)d_ws;
  auto alloc = [&](size_t bytes) -> char* {
    char* r = p;
    p += (bytes + 255) & ~(size_t)255;
    return r;
  };
  __bf16* hb = (__bf16*)alloc((size_t)S_H * DIM * 2);
  __bf16* eb = (__bf16*)alloc((size_t)S_E * DIM * 2);
  __bf16* Wb[8]; for (int i = 0; i < 8; ++i) Wb[i] = (__bf16*)alloc((size_t)DIM * DIM * 2);
  __bf16* db[3]; for (int i = 0; i < 3; ++i) db[i] = (__bf16*)alloc((size_t)RANK * DIM * 2);
  __bf16* ub[3]; for (int i = 0; i < 3; ++i) ub[i] = (__bf16*)alloc((size_t)DIM * RANK * 2);
  float* qf  = (float*)alloc((size_t)S_H * DIM * 4);
  float* kf  = (float*)alloc((size_t)S_H * DIM * 4);
  float* vf  = (float*)alloc((size_t)S_H * DIM * 4);
  float* eqf = (float*)alloc((size_t)S_E * DIM * 4);
  float* ekf = (float*)alloc((size_t)S_E * DIM * 4);
  float* evf = (float*)alloc((size_t)S_E * DIM * 4);
  float*  midf = (float*)alloc((size_t)COND * RANK * 4);
  __bf16* midb = (__bf16*)alloc((size_t)COND * RANK * 2);
  __bf16* Qb = (__bf16*)alloc((size_t)HEADS * S_ALL * HD * 2);
  __bf16* Kb = (__bf16*)alloc((size_t)HEADS * S_ALL * HD * 2);
  __bf16* Vb = (__bf16*)alloc((size_t)HEADS * S_ALL * HD * 2);
  __bf16* Ob = (__bf16*)alloc((size_t)S_ALL * DIM * 2);

  auto conv = [&](const float* src, __bf16* dst, long long n) {
    long long q = (n + 3) / 4;
    int blocks = (int)((q + 255) / 256);
    f32_to_bf16_kernel<<<blocks, 256, 0, stream>>>(src, dst, n);
  };
  auto gemm = [&](const __bf16* A, int lda, const __bf16* B, int ldb,
                  float* C, int ldc, const float* bias,
                  int M, int N, int K, float scale, const float* sptr, int acc) {
    dim3 g((N + BN - 1) / BN, (M + BM - 1) / BM);
    bool aligned = (M % BM == 0) && (N % BN == 0);
    if (aligned)
      gemm_bf16_kernel<false><<<g, 256, 0, stream>>>(A, lda, B, ldb, C, ldc, bias, M, N, K, scale, sptr, acc);
    else
      gemm_bf16_kernel<true><<<g, 256, 0, stream>>>(A, lda, B, ldb, C, ldc, bias, M, N, K, scale, sptr, acc);
  };

  // 1) precision conversion
  conv(hidden,  hb, (long long)S_H * DIM);
  conv(encoder, eb, (long long)S_E * DIM);
  for (int i = 0; i < 8; ++i) conv(W[i], Wb[i], (long long)DIM * DIM);
  for (int i = 0; i < 3; ++i) conv(down[i], db[i], (long long)RANK * DIM);
  for (int i = 0; i < 3; ++i) conv(up[i],   ub[i], (long long)DIM * RANK);

  // 2) base projections (hidden + encoder)
  gemm(hb, DIM, Wb[0], DIM, qf, DIM, bv[0], S_H, DIM, DIM, 1.0f, nullptr, 0);
  gemm(hb, DIM, Wb[1], DIM, kf, DIM, bv[1], S_H, DIM, DIM, 1.0f, nullptr, 0);
  gemm(hb, DIM, Wb[2], DIM, vf, DIM, bv[2], S_H, DIM, DIM, 1.0f, nullptr, 0);
  gemm(eb, DIM, Wb[3], DIM, eqf, DIM, bv[3], S_E, DIM, DIM, 1.0f, nullptr, 0);
  gemm(eb, DIM, Wb[4], DIM, ekf, DIM, bv[4], S_E, DIM, DIM, 1.0f, nullptr, 0);
  gemm(eb, DIM, Wb[5], DIM, evf, DIM, bv[5], S_E, DIM, DIM, 1.0f, nullptr, 0);

  // 3) LoRA deltas on the masked row-window [S_H-COND, S_H) (ALPHA/RANK == 1.0)
  float* tgt[3] = { qf, kf, vf };
  const __bf16* hwin = hb + (size_t)(S_H - COND) * DIM;
  for (int t = 0; t < 3; ++t) {
    gemm(hwin, DIM, db[t], DIM, midf, RANK, nullptr, COND, RANK, DIM, 1.0f, nullptr, 0);
    conv(midf, midb, (long long)COND * RANK);
    gemm(midb, RANK, ub[t], RANK, tgt[t] + (size_t)(S_H - COND) * DIM, DIM, nullptr,
         COND, DIM, RANK, 1.0f, lora_w, 1);
  }

  // 4) heads + RMSNorm + RoPE + concat -> bf16 [H,S,HD]
  qkv_pack_kernel<<<dim3(S_ALL, HEADS), 128, 0, stream>>>(
      qf, kf, vf, eqf, ekf, evf, norm_q, norm_k, norm_aq, norm_ak, ropeC, ropeS, Qb, Kb, Vb);

  // 5) flash attention
  attn_kernel<<<dim3(S_ALL / AT_BQ, HEADS), 128, 0, stream>>>(Qb, Kb, Vb, Ob);

  // 6) output projections straight into d_out (hid first, then enc)
  gemm(Ob + (size_t)S_E * DIM, DIM, Wb[6], DIM, out_hid, DIM, bv[6], S_H, DIM, DIM, 1.0f, nullptr, 0);
  gemm(Ob, DIM, Wb[7], DIM, out_enc, DIM, bv[7], S_E, DIM, DIM, 1.0f, nullptr, 0);
}
